// DiT_67070209294660
// MI455X (gfx1250) — matrix operand order repaired and checked by reading, not compiled, not run on hardware
//
#include <hip/hip_runtime.h>

// ---------------------------------------------------------------------------
// DiT block (spatial + temporal attention with linear-memory) for gfx1250.
// Heavy GEMMs: v_wmma_f32_16x16x32_bf16, 32x64 per-wave tiles (B-frag reuse).
// Attention tile staging: Tensor Data Mover (tensor_load_to_lds) -> LDS.
// LN / RoPE / softmax / elu-memory / gating stay f32 VALU for accuracy.
// ---------------------------------------------------------------------------

typedef __attribute__((ext_vector_type(16))) __bf16 v16bf;
typedef __attribute__((ext_vector_type(8)))  __bf16 v8bf;
typedef __attribute__((ext_vector_type(8)))  float  v8f;
typedef __attribute__((ext_vector_type(4)))  unsigned int u32x4;
typedef __attribute__((ext_vector_type(4)))  int i32x4;
typedef __attribute__((ext_vector_type(8)))  int i32x8;

#define DEV __device__ __forceinline__

static constexpr int B_ = 2, T_ = 16, H_ = 9, W_ = 16;
static constexpr int D_ = 1024, HEADS_ = 16, DH_ = 64, STRIDE_ = 5;
static constexpr int HW_ = H_ * W_;          // 144
static constexpr int N_ = B_ * T_ * HW_;     // 4608
static constexpr int SIXD_ = 6 * D_;         // 6144

DEV unsigned short f2bf(float f) {
  unsigned int u = __float_as_uint(f);
  u += 0x7FFFu + ((u >> 16) & 1u);
  return (unsigned short)(u >> 16);
}
DEV float elu1(float x) { return x > 0.f ? x + 1.f : __expf(x); }
DEV float sigmoidf_(float x) { return 1.f / (1.f + __expf(-x)); }
DEV float gelu_tanh(float x) {
  float x3 = x * x * x;
  return 0.5f * x * (1.f + tanhf(0.7978845608028654f * (x + 0.044715f * x3)));
}

// ---------------- Tensor Data Mover: 2D tile global -> LDS ------------------
// D# per CDNA5 ISA (group0: count/lds/global/type,
// group1: data_size=4B, tensor_dim0/1, tile_dim0/1, dim0 stride).
// This toolchain's builtin is the 6-arg variant (g0, g1, g2, g3, g_extra, cpol).
DEV void tdm_load_2d(const void* lds_dst, const void* gsrc,
                     unsigned tile_w, unsigned tile_h, unsigned stride_elems) {
  unsigned long long ga = (unsigned long long)gsrc;
  unsigned ldso = (unsigned)(unsigned long long)lds_dst;  // low 32b = LDS offset
  u32x4 g0;
  g0[0] = 1u;                                             // count=1, user D#
  g0[1] = ldso;                                           // lds_addr (bytes)
  g0[2] = (unsigned)ga;                                   // global_addr[31:0]
  g0[3] = (unsigned)((ga >> 32) & 0x01FFFFFFu) | (2u << 30);  // addr[56:32], type=2
  i32x8 g1;
  g1[0] = 0x00020000;                                     // data_size = 4 bytes
  g1[1] = (int)((tile_w & 0xFFFFu) << 16);                // tensor_dim0[15:0]
  g1[2] = (int)(((tile_w >> 16) & 0xFFFFu) | ((tile_h & 0xFFFFu) << 16));
  g1[3] = (int)(((tile_h >> 16) & 0xFFFFu) | ((tile_w & 0xFFFFu) << 16)); // tile_dim0
  g1[4] = (int)(tile_h & 0xFFFFu);                        // tile_dim1 (tile_dim2=0)
  g1[5] = (int)stride_elems;                              // tensor_dim0_stride lo32
  g1[6] = 0;
  g1[7] = 0;
  i32x4 gz = {0, 0, 0, 0};
  i32x8 gz8 = {0, 0, 0, 0, 0, 0, 0, 0};
  __builtin_amdgcn_tensor_load_to_lds(g0, g1, gz, gz, gz8, 0);
}

// ---------------- weight convert + transpose: W[K][M] f32 -> Wt[M][K] bf16 --
__global__ void k_convT(const float* __restrict__ w, unsigned short* __restrict__ wt,
                        int K, int M) {
  long long i = (long long)blockIdx.x * 256 + threadIdx.x;
  long long tot = (long long)K * M;
  if (i >= tot) return;
  int k = (int)(i / M), m = (int)(i % M);
  wt[(long long)m * K + k] = f2bf(w[i]);
}

// ---------------- silu(c) ---------------------------------------------------
__global__ void k_silu(const float* __restrict__ c, float* __restrict__ o, int n) {
  int i = blockIdx.x * 256 + threadIdx.x;
  if (i < n) { float v = c[i]; o[i] = v * sigmoidf_(v); }
}

// ---------------- ada modulation: ms = silu(c) @ w_ada + b_ada (32x6144) ----
__global__ void k_ada(const float* __restrict__ sc, const float* __restrict__ w,
                      const float* __restrict__ b, float* __restrict__ ms) {
  int col = blockIdx.x * 256 + threadIdx.x;
  int row = blockIdx.y;
  if (col >= SIXD_) return;
  float acc = b[col];
  const float* cr = sc + (long long)row * D_;
  for (int k = 0; k < D_; ++k) acc += cr[k] * w[(long long)k * SIXD_ + col];
  ms[(long long)row * SIXD_ + col] = acc;
}

// ---------------- LN + adaLN modulate -> bf16 -------------------------------
__global__ void k_lnmod(const float* __restrict__ x, const float* __restrict__ ms,
                        int shChunk, int scChunk, unsigned short* __restrict__ out) {
  __shared__ float red[256];
  __shared__ float s_mean, s_rstd;
  int tok = blockIdx.x, tid = threadIdx.x;
  const float* xr = x + (long long)tok * D_;
  float v[4];
  float s = 0.f;
  for (int i = 0; i < 4; ++i) { v[i] = xr[tid + 256 * i]; s += v[i]; }
  red[tid] = s; __syncthreads();
  for (int o = 128; o > 0; o >>= 1) { if (tid < o) red[tid] += red[tid + o]; __syncthreads(); }
  if (tid == 0) s_mean = red[0] / (float)D_;
  __syncthreads();
  float m = s_mean;
  s = 0.f;
  for (int i = 0; i < 4; ++i) { float d = v[i] - m; s += d * d; }
  red[tid] = s; __syncthreads();
  for (int o = 128; o > 0; o >>= 1) { if (tid < o) red[tid] += red[tid + o]; __syncthreads(); }
  if (tid == 0) s_rstd = rsqrtf(red[0] / (float)D_ + 1e-6f);
  __syncthreads();
  float r = s_rstd;
  int bt = tok / HW_;
  const float* msr = ms + (long long)bt * SIXD_;
  for (int i = 0; i < 4; ++i) {
    int d = tid + 256 * i;
    float y = (v[i] - m) * r;
    y = y * (1.f + msr[scChunk * D_ + d]) + msr[shChunk * D_ + d];
    out[(long long)tok * D_ + d] = f2bf(y);
  }
}

// ---------------- WMMA bf16 GEMM: C[N][M] = A[N][K] * Wt[M][K]^T ------------
// Block 256 thr = 8 waves; wave computes 32 rows x 64 cols (8 wmma accums,
// each B fragment reused for 2 A fragments).
// EPI: 0 = store, 1 = +bias, 2 = +bias then gelu.  OUT_BF: store bf16.
template <int EPI, bool OUT_BF>
__global__ void k_gemm(const __bf16* __restrict__ A, const __bf16* __restrict__ Wt,
                       const float* __restrict__ bias,
                       float* __restrict__ outF, unsigned short* __restrict__ outB,
                       int Nrows, int M, int K) {
  int wave = threadIdx.x >> 5;
  int lane = threadIdx.x & 31;
  int lgrp = lane >> 4;          // which 16-lane half
  int lrow = lane & 15;
  int m0 = (blockIdx.y * 8 + wave) * 32;
  int n0 = blockIdx.x * 64;
  if (m0 >= Nrows) return;

  v8f acc[8] = {};
  const __bf16* arow0 = A + (long long)(m0 + lrow) * K + 16 * lgrp;
  const __bf16* arow1 = arow0 + (long long)16 * K;
  const __bf16* bp0 = Wt + (long long)(n0 +  0 + lrow) * K + 16 * lgrp;
  const __bf16* bp1 = Wt + (long long)(n0 + 16 + lrow) * K + 16 * lgrp;
  const __bf16* bp2 = Wt + (long long)(n0 + 32 + lrow) * K + 16 * lgrp;
  const __bf16* bp3 = Wt + (long long)(n0 + 48 + lrow) * K + 16 * lgrp;

  for (int k = 0; k < K; k += 32) {
    // A fragments (16x32 bf16 lane layout): K = k + 16*lgrp + {0..15}.
    v8bf a0lo = *(const v8bf*)(arow0 + k);
    v8bf a0hi = *(const v8bf*)(arow0 + k + 8);
    v8bf a1lo = *(const v8bf*)(arow1 + k);
    v8bf a1hi = *(const v8bf*)(arow1 + k + 8);
    v16bf a0, a1;
#pragma unroll
    for (int i = 0; i < 8; ++i) {
      a0[i] = a0lo[i]; a0[i + 8] = a0hi[i];
      a1[i] = a1lo[i]; a1[i + 8] = a1hi[i];
    }
    // B fragments (32x16): lane col = lrow, K = k + 16*lgrp + {0..15}.
    v16bf b;
    b = *(const v16bf*)(bp0 + k);
    acc[0] = __builtin_amdgcn_wmma_f32_16x16x32_bf16(false, a0, false, b, (short)0, acc[0], false, false);
    acc[4] = __builtin_amdgcn_wmma_f32_16x16x32_bf16(false, a1, false, b, (short)0, acc[4], false, false);
    b = *(const v16bf*)(bp1 + k);
    acc[1] = __builtin_amdgcn_wmma_f32_16x16x32_bf16(false, a0, false, b, (short)0, acc[1], false, false);
    acc[5] = __builtin_amdgcn_wmma_f32_16x16x32_bf16(false, a1, false, b, (short)0, acc[5], false, false);
    b = *(const v16bf*)(bp2 + k);
    acc[2] = __builtin_amdgcn_wmma_f32_16x16x32_bf16(false, a0, false, b, (short)0, acc[2], false, false);
    acc[6] = __builtin_amdgcn_wmma_f32_16x16x32_bf16(false, a1, false, b, (short)0, acc[6], false, false);
    b = *(const v16bf*)(bp3 + k);
    acc[3] = __builtin_amdgcn_wmma_f32_16x16x32_bf16(false, a0, false, b, (short)0, acc[3], false, false);
    acc[7] = __builtin_amdgcn_wmma_f32_16x16x32_bf16(false, a1, false, b, (short)0, acc[7], false, false);
  }

#pragma unroll
  for (int hh = 0; hh < 2; ++hh) {
#pragma unroll
    for (int j = 0; j < 4; ++j) {
#pragma unroll
      for (int r = 0; r < 8; ++r) {
        int row = m0 + hh * 16 + 8 * lgrp + r;  // C/D layout: VGPR r -> M = r + 8*lgrp
        int col = n0 + j * 16 + lrow;           // N = lane%16
        float val = acc[hh * 4 + j][r];
        if (EPI >= 1) val += bias[col];
        if (EPI == 2) val = gelu_tanh(val);
        long long idx = (long long)row * M + col;
        if (OUT_BF) outB[idx] = f2bf(val);
        else        outF[idx] = val;
      }
    }
  }
}

// ---------------- spatial pixel-RoPE frequency ------------------------------
DEV float pix_freq(int p, int d) {
  const float PI = 3.14159265358979323846f;
  if (d < DH_ / 2) {
    int j = d >> 1;
    float fr = (1.0f + 127.0f * (float)j / 15.0f) * PI;     // linspace(1,128,16)*pi
    float pos = -1.0f + 2.0f * (float)(p / W_) / (float)(H_ - 1);
    return pos * fr;
  } else {
    int j = (d - DH_ / 2) >> 1;
    float fr = (1.0f + 127.0f * (float)j / 15.0f) * PI;
    float pos = -1.0f + 2.0f * (float)(p % W_) / (float)(W_ - 1);
    return pos * fr;
  }
}

// ---------------- spatial attention (one block per (bt, head)) --------------
__global__ void k_attn_s(const float* __restrict__ qkv,
                         const float* __restrict__ mem_in, const float* __restrict__ z_in,
                         const float* __restrict__ beta,
                         unsigned short* __restrict__ attno,
                         float* __restrict__ mem_out, float* __restrict__ z_out) {
  extern __shared__ float sm[];
  float* sk   = sm;                         // 144*64
  float* sv   = sk + HW_ * DH_;             // 144*64
  float* smem = sv + HW_ * DH_;             // 64*64
  float* sz   = smem + DH_ * DH_;           // 64
  float* qrow = sz + DH_;                   // 64
  float* eq   = qrow + DH_;                 // 64
  float* srow = eq + DH_;                   // 144

  int head = blockIdx.x & 15;
  int bt = blockIdx.x >> 4;
  int tid = threadIdx.x;
  long long rowBase = (long long)bt * HW_;
  int colq = head * DH_, colk = D_ + head * DH_, colv = 2 * D_ + head * DH_;
  long long mh = (long long)(bt * HEADS_ + head);

  // TDM: DMA k, v, mem, z tiles into LDS (issued once, by wave 0).
  if (tid < 32) {
    tdm_load_2d(sk, qkv + rowBase * (3 * D_) + colk, DH_, HW_, 3 * D_);
    tdm_load_2d(sv, qkv + rowBase * (3 * D_) + colv, DH_, HW_, 3 * D_);
    tdm_load_2d(smem, mem_in + mh * DH_ * DH_, DH_ * DH_, 1, DH_ * DH_);
    tdm_load_2d(sz, z_in + mh * DH_, DH_, 1, DH_);
    __builtin_amdgcn_s_wait_tensorcnt(0);
  }
  __syncthreads();

  // memory update (pre-RoPE k, v): new_z = z + sum elu1(k); new_mem += elu1(elu1(k[:5])) x v[:5]
  for (int d = tid; d < DH_; d += 256) {
    float acc = sz[d];
    for (int p = 0; p < HW_; ++p) acc += elu1(sk[p * DH_ + d]);
    z_out[mh * DH_ + d] = acc;
  }
  for (int i = tid; i < DH_ * DH_; i += 256) {
    int d = i / DH_, e = i % DH_;
    float acc = smem[i];
    for (int p = 0; p < STRIDE_; ++p) acc += elu1(elu1(sk[p * DH_ + d])) * sv[p * DH_ + e];
    mem_out[mh * DH_ * DH_ + i] = acc;
  }
  __syncthreads();

  // RoPE k in LDS (pairwise, in place)
  for (int pr = tid; pr < HW_ * (DH_ / 2); pr += 256) {
    int p = pr / (DH_ / 2), d = (pr % (DH_ / 2)) * 2;
    float f = pix_freq(p, d);
    float cs = __cosf(f), sn = __sinf(f);
    float x0 = sk[p * DH_ + d], x1 = sk[p * DH_ + d + 1];
    sk[p * DH_ + d]     = x0 * cs - x1 * sn;
    sk[p * DH_ + d + 1] = x1 * cs + x0 * sn;
  }
  __syncthreads();

  float g = sigmoidf_(beta[head]);
  const float scale = 0.125f;  // 1/sqrt(64)

  for (int i = 0; i < HW_; ++i) {
    if (tid < DH_ / 2) {
      int d = tid * 2;
      const float* rp = qkv + (rowBase + i) * (3 * D_) + colq;
      float f = pix_freq(i, d);
      float cs = __cosf(f), sn = __sinf(f);
      float x0 = rp[d], x1 = rp[d + 1];
      float y0 = x0 * cs - x1 * sn, y1 = x1 * cs + x0 * sn;
      qrow[d] = y0; qrow[d + 1] = y1;
      eq[d] = elu1(y0); eq[d + 1] = elu1(y1);
    }
    __syncthreads();
    if (tid < HW_) {
      float acc = 0.f;
      const float* kp = sk + tid * DH_;
      for (int d = 0; d < DH_; ++d) acc += qrow[d] * kp[d];
      srow[tid] = acc * scale;
    }
    __syncthreads();
    if (tid < DH_) {
      int e = tid;
      float mx = -1e30f;
      for (int j = 0; j < HW_; ++j) mx = fmaxf(mx, srow[j]);
      float den = 0.f, acc = 0.f;
      for (int j = 0; j < HW_; ++j) {
        float pj = __expf(srow[j] - mx);
        den += pj; acc += pj * sv[j * DH_ + e];
      }
      float o = acc / den;
      float num = 0.f, rden = 0.f;
      for (int d = 0; d < DH_; ++d) { num += eq[d] * smem[d * DH_ + e]; rden += eq[d] * sz[d]; }
      float res = g * (num / rden) + (1.f - g) * o;
      attno[(rowBase + i) * D_ + head * DH_ + e] = f2bf(res);
    }
    __syncthreads();
  }
}

// ---------------- temporal attention (one block per (b*HW + hw, head)) ------
__global__ void k_attn_t(const float* __restrict__ qkv,
                         const float* __restrict__ mem_in, const float* __restrict__ z_in,
                         const float* __restrict__ beta,
                         unsigned short* __restrict__ attno,
                         float* __restrict__ mem_out, float* __restrict__ z_out) {
  __shared__ float sq[T_ * DH_], sk[T_ * DH_], sv[T_ * DH_];
  __shared__ float smem[DH_ * DH_], sz[DH_], srow[T_];
  int head = blockIdx.x & 15;
  int gidx = blockIdx.x >> 4;            // b*HW + h*W + w
  int b = gidx / HW_, hw = gidx % HW_;
  int tid = threadIdx.x;                 // 128 threads
  long long mh = (long long)gidx * HEADS_ + head;

  // TDM: rows are T_ time steps strided by HW_*3D floats in the qkv buffer.
  if (tid < 32) {
    long long base = (long long)(b * T_ * HW_ + hw) * (3 * D_);
    tdm_load_2d(sq, qkv + base + head * DH_,          DH_, T_, HW_ * 3 * D_);
    tdm_load_2d(sk, qkv + base + D_ + head * DH_,     DH_, T_, HW_ * 3 * D_);
    tdm_load_2d(sv, qkv + base + 2 * D_ + head * DH_, DH_, T_, HW_ * 3 * D_);
    tdm_load_2d(smem, mem_in + mh * DH_ * DH_, DH_ * DH_, 1, DH_ * DH_);
    tdm_load_2d(sz, z_in + mh * DH_, DH_, 1, DH_);
    __builtin_amdgcn_s_wait_tensorcnt(0);
  }
  __syncthreads();

  for (int d = tid; d < DH_; d += 128) {
    float acc = sz[d];
    for (int t = 0; t < T_; ++t) acc += elu1(sk[t * DH_ + d]);
    z_out[mh * DH_ + d] = acc;
  }
  for (int i = tid; i < DH_ * DH_; i += 128) {
    int d = i / DH_, e = i % DH_;
    float acc = smem[i];
    for (int t = 0; t < STRIDE_; ++t) acc += elu1(elu1(sk[t * DH_ + d])) * sv[t * DH_ + e];
    mem_out[mh * DH_ * DH_ + i] = acc;
  }
  __syncthreads();

  // lang RoPE on q and k: f(t,d) = t * 10000^(-(d&~1)/64)
  for (int pr = tid; pr < T_ * (DH_ / 2); pr += 128) {
    int t = pr / (DH_ / 2), d = (pr % (DH_ / 2)) * 2;
    float f = (float)t * __powf(10000.f, -(float)d / (float)DH_);
    float cs = __cosf(f), sn = __sinf(f);
    float q0 = sq[t * DH_ + d], q1 = sq[t * DH_ + d + 1];
    sq[t * DH_ + d] = q0 * cs - q1 * sn;
    sq[t * DH_ + d + 1] = q1 * cs + q0 * sn;
    float k0 = sk[t * DH_ + d], k1 = sk[t * DH_ + d + 1];
    sk[t * DH_ + d] = k0 * cs - k1 * sn;
    sk[t * DH_ + d + 1] = k1 * cs + k0 * sn;
  }
  __syncthreads();

  float g = sigmoidf_(beta[head]);
  for (int i = 0; i < T_; ++i) {
    if (tid <= i) {  // causal: keys j <= i
      float acc = 0.f;
      for (int d = 0; d < DH_; ++d) acc += sq[i * DH_ + d] * sk[tid * DH_ + d];
      srow[tid] = acc * 0.125f;
    }
    __syncthreads();
    if (tid < DH_) {
      int e = tid;
      float mx = -1e30f;
      for (int j = 0; j <= i; ++j) mx = fmaxf(mx, srow[j]);
      float den = 0.f, acc = 0.f;
      for (int j = 0; j <= i; ++j) {
        float pj = __expf(srow[j] - mx);
        den += pj; acc += pj * sv[j * DH_ + e];
      }
      float o = acc / den;
      float num = 0.f, rden = 0.f;
      for (int d = 0; d < DH_; ++d) {
        float eqv = elu1(sq[i * DH_ + d]);
        num += eqv * smem[d * DH_ + e]; rden += eqv * sz[d];
      }
      float res = g * (num / rden) + (1.f - g) * o;
      long long row = (long long)((b * T_ + i) * HW_ + hw);
      attno[row * D_ + head * DH_ + e] = f2bf(res);
    }
    __syncthreads();
  }
}

// ---------------- gating / residual elementwise -----------------------------
__global__ void k_gate_mul1p(const float* __restrict__ xa, const float* __restrict__ ms,
                             int chunk, float* __restrict__ dst) {
  long long i = (long long)blockIdx.x * 256 + threadIdx.x;
  if (i >= (long long)N_ * D_) return;
  int ch = (int)(i % D_);
  int bt = (int)(i / D_) / HW_;
  float gv = ms[(long long)bt * SIXD_ + chunk * D_ + ch];
  dst[i] = xa[i] * (1.f + gv);
}
__global__ void k_gate_addmul(const float* __restrict__ src, const float* __restrict__ y,
                              const float* __restrict__ ms, int chunk,
                              float* __restrict__ dst) {
  long long i = (long long)blockIdx.x * 256 + threadIdx.x;
  if (i >= (long long)N_ * D_) return;
  int ch = (int)(i % D_);
  int bt = (int)(i / D_) / HW_;
  float gv = ms[(long long)bt * SIXD_ + chunk * D_ + ch];
  dst[i] = src[i] + y[i] * gv;
}

// ---------------------------------------------------------------------------
extern "C" void kernel_launch(void* const* d_in, const int* in_sizes, int n_in,
                              void* d_out, int out_size, void* d_ws, size_t ws_size,
                              hipStream_t stream) {
  (void)in_sizes; (void)n_in; (void)out_size; (void)ws_size;
  const float* x       = (const float*)d_in[0];
  const float* c       = (const float*)d_in[1];
  const float* mem_s   = (const float*)d_in[2];
  const float* z_s     = (const float*)d_in[3];
  const float* mem_t   = (const float*)d_in[4];
  const float* z_t     = (const float*)d_in[5];
  const float* w_qkv_s = (const float*)d_in[6];
  const float* w_out_s = (const float*)d_in[7];
  const float* b_out_s = (const float*)d_in[8];
  const float* beta_s  = (const float*)d_in[9];
  const float* w_ada_s = (const float*)d_in[10];
  const float* b_ada_s = (const float*)d_in[11];
  const float* w1_s    = (const float*)d_in[12];
  const float* b1_s    = (const float*)d_in[13];
  const float* w2_s    = (const float*)d_in[14];
  const float* b2_s    = (const float*)d_in[15];
  const float* w_qkv_t = (const float*)d_in[16];
  const float* w_out_t = (const float*)d_in[17];
  const float* b_out_t = (const float*)d_in[18];
  const float* beta_t  = (const float*)d_in[19];
  const float* w_ada_t = (const float*)d_in[20];
  const float* b_ada_t = (const float*)d_in[21];
  const float* w1_t    = (const float*)d_in[22];
  const float* b1_t    = (const float*)d_in[23];
  const float* w2_t    = (const float*)d_in[24];
  const float* b2_t    = (const float*)d_in[25];

  // workspace layout (bytes, 64B aligned)
  const size_t O_WQKV_S = 0;            // 3072x1024 bf16
  const size_t O_WOUT_S = 6291456;      // 1024x1024 bf16
  const size_t O_W1_S   = 8388608;      // 4096x1024 bf16
  const size_t O_W2_S   = 16777216;     // 1024x4096 bf16
  const size_t O_WQKV_T = 25165824;
  const size_t O_WOUT_T = 31457280;
  const size_t O_W1_T   = 33554432;
  const size_t O_W2_T   = 41943040;
  const size_t O_SILUC  = 50331648;     // 32x1024 f32
  const size_t O_MS_S   = 50462720;     // 32x6144 f32
  const size_t O_MS_T   = 51249152;
  const size_t O_XMOD   = 52035584;     // 4608x1024 bf16
  const size_t O_QKV    = 61472768;     // 4608x3072 f32
  const size_t O_ATTNO  = 118095872;    // 4608x1024 bf16
  const size_t O_XA     = 127533056;    // 4608x1024 f32
  const size_t O_XCUR   = 146407424;    // 4608x1024 f32
  const size_t O_HBF    = 165281792;    // 4608x4096 bf16
  const size_t O_Y      = 203030528;    // 4608x1024 f32

  char* ws = (char*)d_ws;
  auto F  = [&](size_t off) { return (float*)(ws + off); };
  auto U  = [&](size_t off) { return (unsigned short*)(ws + off); };
  auto BF = [&](size_t off) { return (const __bf16*)(ws + off); };

  float* out = (float*)d_out;
  float* out_mem_s = out + 4718592;
  float* out_z_s   = out + 6815744;
  float* out_mem_t = out + 6848512;
  float* out_z_t   = out + 25722880;

  // 1) convert + transpose weights to bf16 [M][K]
  auto convT = [&](const float* w, size_t dst, int K, int M) {
    long long tot = (long long)K * M;
    k_convT<<<(unsigned)((tot + 255) / 256), 256, 0, stream>>>(w, U(dst), K, M);
  };
  convT(w_qkv_s, O_WQKV_S, 1024, 3072);
  convT(w_out_s, O_WOUT_S, 1024, 1024);
  convT(w1_s,    O_W1_S,   1024, 4096);
  convT(w2_s,    O_W2_S,   4096, 1024);
  convT(w_qkv_t, O_WQKV_T, 1024, 3072);
  convT(w_out_t, O_WOUT_T, 1024, 1024);
  convT(w1_t,    O_W1_T,   1024, 4096);
  convT(w2_t,    O_W2_T,   4096, 1024);

  // 2) silu(c), ada modulation for both branches
  k_silu<<<(32 * 1024 + 255) / 256, 256, 0, stream>>>(c, F(O_SILUC), 32 * 1024);
  {
    dim3 g(SIXD_ / 256, 32);
    k_ada<<<g, 256, 0, stream>>>(F(O_SILUC), w_ada_s, b_ada_s, F(O_MS_S));
    k_ada<<<g, 256, 0, stream>>>(F(O_SILUC), w_ada_t, b_ada_t, F(O_MS_T));
  }

  auto gemm = [&](int epi, bool outbf, size_t aOff, size_t wOff, const float* bias,
                  void* o, int Nr, int M, int K) {
    dim3 g(M / 64, Nr / 256), b(256);
    if (epi == 0)
      k_gemm<0, false><<<g, b, 0, stream>>>(BF(aOff), BF(wOff), bias, (float*)o, nullptr, Nr, M, K);
    else if (epi == 1)
      k_gemm<1, false><<<g, b, 0, stream>>>(BF(aOff), BF(wOff), bias, (float*)o, nullptr, Nr, M, K);
    else if (outbf)
      k_gemm<2, true><<<g, b, 0, stream>>>(BF(aOff), BF(wOff), bias, nullptr, (unsigned short*)o, Nr, M, K);
  };

  const int EW_GRID = (int)(((long long)N_ * D_ + 255) / 256);
  const size_t SH_ATTN_S = (size_t)(2 * HW_ * DH_ + DH_ * DH_ + 3 * DH_ + HW_) * sizeof(float);

  // ---------------- spatial branch ----------------
  k_lnmod<<<N_, 256, 0, stream>>>(x, F(O_MS_S), 0, 1, U(O_XMOD));
  gemm(0, false, O_XMOD, O_WQKV_S, nullptr, F(O_QKV), N_, 3 * D_, D_);
  k_attn_s<<<B_ * T_ * HEADS_, 256, SH_ATTN_S, stream>>>(
      F(O_QKV), mem_s, z_s, beta_s, U(O_ATTNO), out_mem_s, out_z_s);
  gemm(1, false, O_ATTNO, O_WOUT_S, b_out_s, F(O_XA), N_, D_, D_);
  k_gate_mul1p<<<EW_GRID, 256, 0, stream>>>(F(O_XA), F(O_MS_S), 2, F(O_XCUR));
  k_lnmod<<<N_, 256, 0, stream>>>(F(O_XCUR), F(O_MS_S), 3, 4, U(O_XMOD));
  gemm(2, true, O_XMOD, O_W1_S, b1_s, U(O_HBF), N_, 4096, D_);
  gemm(1, false, O_HBF, O_W2_S, b2_s, F(O_Y), N_, D_, 4096);
  k_gate_addmul<<<EW_GRID, 256, 0, stream>>>(F(O_XCUR), F(O_Y), F(O_MS_S), 5, F(O_XCUR));

  // ---------------- temporal branch ----------------
  k_lnmod<<<N_, 256, 0, stream>>>(F(O_XCUR), F(O_MS_T), 0, 1, U(O_XMOD));
  gemm(0, false, O_XMOD, O_WQKV_T, nullptr, F(O_QKV), N_, 3 * D_, D_);
  k_attn_t<<<B_ * HW_ * HEADS_, 128, 0, stream>>>(
      F(O_QKV), mem_t, z_t, beta_t, U(O_ATTNO), out_mem_t, out_z_t);
  gemm(1, false, O_ATTNO, O_WOUT_T, b_out_t, F(O_XA), N_, D_, D_);
  k_gate_mul1p<<<EW_GRID, 256, 0, stream>>>(F(O_XA), F(O_MS_T), 2, F(O_XCUR));
  k_lnmod<<<N_, 256, 0, stream>>>(F(O_XCUR), F(O_MS_T), 3, 4, U(O_XMOD));
  gemm(2, true, O_XMOD, O_W1_T, b1_t, U(O_HBF), N_, 4096, D_);
  gemm(1, false, O_HBF, O_W2_T, b2_t, F(O_Y), N_, D_, 4096);
  // final residual written straight into d_out's x region
  k_gate_addmul<<<EW_GRID, 256, 0, stream>>>(F(O_XCUR), F(O_Y), F(O_MS_T), 5, out);
}
